// CausalSelfAttention_74122545594589
// MI455X (gfx1250) — compile-verified, matrix operand order checked by reading, not run on hardware
//
#include <hip/hip_runtime.h>

// ---------------- problem constants ----------------
#define S_LEN  2048
#define D_DIM  2048
#define NHEAD  16
#define HD     128
#define BATCH  2
#define M_ROWS (BATCH * S_LEN) // 4096

// GEMM tiling: block = 256 threads (8 waves); block tile 256(M) x 128(N);
// each wave computes 32(M) x 128(N); K-step 32, double-buffered async staging.
#define BM 256
#define BN 128
#define BK 32
#define LDS_ROW 40  // bf16 elems per LDS row: 32 data + 8 pad (bank-conflict free)

typedef __bf16 bf16_t;
typedef __attribute__((ext_vector_type(16))) __bf16 v16bf;
typedef __attribute__((ext_vector_type(8)))  __bf16 v8bf;
typedef __attribute__((ext_vector_type(8)))  float  v8f;

// D = A(16x32 bf16) * B(32x16 bf16) + C(16x16 f32)
__device__ __forceinline__ v8f wmma_bf16(v16bf a, v16bf b, v8f c) {
  return __builtin_amdgcn_wmma_f32_16x16x32_bf16(false, a, false, b, (short)0, c,
                                                 false, false);
}

// Load a 16x32 bf16 WMMA fragment (A layout; B layout identical with "row"
// meaning the N column) from a row-major source with leading dim `ld` elems.
// Lane r (and r+16) owns row r; K halves {h*8..h*8+7} and {16+h*8..}.
__device__ __forceinline__ v16bf load_frag(const bf16_t* base, int ld) {
  const int lane = threadIdx.x & 31;
  const int r = lane & 15, h = lane >> 4;
  const bf16_t* p = base + (long)r * ld + h * 8;
  v8bf lo = *(const v8bf*)p;
  v8bf hi = *(const v8bf*)(p + 16);
  v16bf f;
#pragma unroll
  for (int i = 0; i < 8; ++i) { f[i] = lo[i]; f[8 + i] = hi[i]; }
  return f;
}

// Per-lane asynchronous global -> LDS copy of 16 bytes (ASYNCcnt-tracked).
__device__ __forceinline__ void async_copy16(unsigned int lds_byte_addr,
                                             const void* gptr) {
  asm volatile("global_load_async_to_lds_b128 %0, %1, off"
               :: "v"(lds_byte_addr), "v"(gptr)
               : "memory");
}

__device__ __forceinline__ void wait_async0() {
  asm volatile("s_wait_asynccnt 0" ::: "memory");
}

// Low 32 bits of a flat pointer to a __shared__ object == LDS byte address.
__device__ __forceinline__ unsigned int lds_addr(const void* p) {
  return (unsigned int)(uintptr_t)p;
}

// ---------------- f32 -> bf16 convert ----------------
__global__ void cvt_kernel(const float* __restrict__ src, bf16_t* __restrict__ dst,
                           int n) {
  int i = blockIdx.x * blockDim.x + threadIdx.x;
  if (i < n) dst[i] = (bf16_t)src[i];
}

// ---------------- LDS-tiled async WMMA GEMM ----------------
// y[m,n] = sum_k A[m,k] * W[n,k].
// mode 0/1: write bf16 [B,H,S,hd]; mode 2: bf16 [B,H,hd,S] (V transposed);
// mode 3: f32 row-major [M, N].
__global__ __launch_bounds__(256) void gemm_lds(
    const bf16_t* __restrict__ A, const bf16_t* __restrict__ W,
    bf16_t* __restrict__ outb, float* __restrict__ outf, int mode) {
  __shared__ bf16_t sA[2][BM * LDS_ROW];  // 2 x 20 KB
  __shared__ bf16_t sB[2][BN * LDS_ROW];  // 2 x 10 KB

  const int tid = threadIdx.x;
  const int wv = tid >> 5;            // wave 0..7
  const int mblk = blockIdx.x >> 4;   // M/BM = 16
  const int nblk = blockIdx.x & 15;   // N/BN = 16
  const int m0 = mblk * BM, n0 = nblk * BN;
  const int K = D_DIM;

  // Stage one K-step: rows x 32 bf16 (64 B = 4 x 16 B chunks per row).
  // A: 1024 chunks (4/thread); B: 512 chunks (2/thread).
  auto stage = [&](int buf, int k0) {
#pragma unroll
    for (int c = tid; c < BM * 4; c += 256) {
      const int row = c >> 2, q = c & 3;
      const unsigned int loff = (unsigned int)(row * LDS_ROW + q * 8) * 2u;
      async_copy16(lds_addr(&sA[buf][0]) + loff,
                   A + (long)(m0 + row) * K + k0 + q * 8);
    }
#pragma unroll
    for (int c = tid; c < BN * 4; c += 256) {
      const int row = c >> 2, q = c & 3;
      const unsigned int loff = (unsigned int)(row * LDS_ROW + q * 8) * 2u;
      async_copy16(lds_addr(&sB[buf][0]) + loff,
                   W + (long)(n0 + row) * K + k0 + q * 8);
    }
  };

  stage(0, 0);

  v8f acc0[8] = {}, acc1[8] = {};
  const int mrow = wv * 32;
  for (int k0 = 0; k0 < K; k0 += BK) {
    const int buf = (k0 >> 5) & 1;
    wait_async0();       // own wave's async fills complete
    __syncthreads();     // everyone's fills complete; prev buffer fully read
    if (k0 + BK < K) stage(buf ^ 1, k0 + BK);

    v16bf a0 = load_frag(&sA[buf][mrow * LDS_ROW], LDS_ROW);
    v16bf a1 = load_frag(&sA[buf][(mrow + 16) * LDS_ROW], LDS_ROW);
    v16bf b = load_frag(&sB[buf][0], LDS_ROW);
#pragma unroll
    for (int t = 0; t < 8; ++t) {
      v16bf bn = b;   // pipeline: fetch tile t+1 before using tile t
      if (t < 7) bn = load_frag(&sB[buf][(t + 1) * 16 * LDS_ROW], LDS_ROW);
      acc0[t] = wmma_bf16(a0, b, acc0[t]);
      acc1[t] = wmma_bf16(a1, b, acc1[t]);
      b = bn;
    }
  }

  const int lane = tid & 31;
  const int r = lane & 15, hb8 = (lane >> 4) * 8;
#pragma unroll
  for (int g = 0; g < 2; ++g) {
#pragma unroll
    for (int t = 0; t < 8; ++t) {
#pragma unroll
      for (int j = 0; j < 8; ++j) {
        const int m = m0 + mrow + g * 16 + hb8 + j;  // global row (b*S + s)
        const int n = n0 + t * 16 + r;               // global col (h*hd + dim)
        const float v = g ? acc1[t][j] : acc0[t][j];
        if (mode == 3) {
          outf[(long)m * D_DIM + n] = v;
        } else {
          const int b = m >> 11, s = m & (S_LEN - 1);
          const int h = n >> 7, dd = n & (HD - 1);
          long idx;
          if (mode == 2)
            idx = ((long)(b * NHEAD + h) * HD + dd) * S_LEN + s;   // V^T
          else
            idx = ((long)(b * NHEAD + h) * S_LEN + s) * HD + dd;   // Q, K
          outb[idx] = (bf16_t)v;
        }
      }
    }
  }
}

// ---------------- RoPE (interleaved pairs) on bf16 [B,H,S,hd] ----------------
__global__ void rope_kernel(bf16_t* __restrict__ T, int npairs) {
  int idx = blockIdx.x * blockDim.x + threadIdx.x;
  if (idx >= npairs) return;
  const int i = idx & 63;                 // pair index, hd/2 = 64
  const int s = (idx >> 6) & (S_LEN - 1); // sequence position
  const float ang = (float)s * __expf(-(float)(2 * i) * (9.210340371976184f / 128.0f));
  float sn, cs;
  __sincosf(ang, &sn, &cs);
  bf16_t* p = T + (long)idx * 2;
  const float x1 = (float)p[0], x2 = (float)p[1];
  p[0] = (bf16_t)(x1 * cs - x2 * sn);
  p[1] = (bf16_t)(x1 * sn + x2 * cs);
}

// ---------------- fused causal flash attention ----------------
// One wave per 16-query tile per (b,h). Scores computed transposed
// (S^T = K * Q^T) so the softmaxed tiles land directly in the A-fragment
// layout of P for the P*V WMMA. Online softmax, f32 accumulators.
// Fragment loads are software-pipelined one tile ahead of their WMMA.
__global__ __launch_bounds__(128) void flash_kernel(
    const bf16_t* __restrict__ Q, const bf16_t* __restrict__ K,
    const bf16_t* __restrict__ Vt, bf16_t* __restrict__ O) {
  const int wave = blockIdx.x * 4 + (threadIdx.x >> 5);
  const int bh = wave >> 7;  // S/16 = 128 query tiles per (b,h)
  const int qt = wave & 127;
  const int q0 = qt * 16;
  const int b = bh >> 4, h = bh & 15;
  const bf16_t* Qp = Q + (long)bh * S_LEN * HD;
  const bf16_t* Kp = K + (long)bh * S_LEN * HD;
  const bf16_t* Vp = Vt + (long)bh * HD * S_LEN;

  const int lane = threadIdx.x & 31;
  const int r = lane & 15, hb8 = (lane >> 4) * 8;
  const float scale = 0.08838834764831845f;  // 1/sqrt(128)
  const float NEG = -3.0e38f;

  v16bf bq[4];
#pragma unroll
  for (int t = 0; t < 4; ++t) bq[t] = load_frag(Qp + (long)q0 * HD + t * 32, HD);

  v8f o[8] = {};
  float mrun = NEG, lrun = 0.0f;
  const float qi = (float)(q0 + r);

  for (int kk = 0; kk < q0 + 16; kk += 32) {
    // ---- scores S^T = K * Q^T (keys on M, queries on N) ----
    v8f s0 = {}, s1 = {};
    v16bf a0 = load_frag(Kp + (long)kk * HD, HD);
    v16bf a1 = load_frag(Kp + (long)(kk + 16) * HD, HD);
#pragma unroll
    for (int t = 0; t < 4; ++t) {
      v16bf na0 = a0, na1 = a1;
      if (t < 3) {
        na0 = load_frag(Kp + (long)kk * HD + (t + 1) * 32, HD);
        na1 = load_frag(Kp + (long)(kk + 16) * HD + (t + 1) * 32, HD);
      }
      s0 = wmma_bf16(a0, bq[t], s0);
      s1 = wmma_bf16(a1, bq[t], s1);
      a0 = na0; a1 = na1;
    }

    // ---- scale + causal mask + online softmax ----
    float mloc = NEG;
#pragma unroll
    for (int j = 0; j < 8; ++j) {
      const float k0f = (float)(kk + hb8 + j);
      const float a  = (k0f <= qi)         ? s0[j] * scale : NEG;
      const float bb = (k0f + 16.0f <= qi) ? s1[j] * scale : NEG;
      s0[j] = a; s1[j] = bb;
      mloc = fmaxf(mloc, fmaxf(a, bb));
    }
    mloc = fmaxf(mloc, __shfl_xor(mloc, 16, 32));
    const float mnew = fmaxf(mrun, mloc);
    const float corr = __expf(mrun - mnew);

    float psum = 0.0f;
    v16bf pa;  // P^T packed directly in A-fragment layout (16q x 32k)
#pragma unroll
    for (int j = 0; j < 8; ++j) {
      const float p0 = __expf(s0[j] - mnew);
      const float p1 = __expf(s1[j] - mnew);
      psum += p0 + p1;
      pa[j] = (bf16_t)p0;
      pa[8 + j] = (bf16_t)p1;
    }
    psum += __shfl_xor(psum, 16, 32);
    lrun = lrun * corr + psum;
    mrun = mnew;

    float cj[8];
#pragma unroll
    for (int j = 0; j < 8; ++j) cj[j] = __shfl(corr, hb8 + j, 32);

    // ---- O = diag(corr)*O + P * V ----
    v16bf bv = load_frag(Vp + (long)kk, S_LEN);  // dim tile 0
#pragma unroll
    for (int t = 0; t < 8; ++t) {
      v16bf nbv = bv;
      if (t < 7) nbv = load_frag(Vp + (long)((t + 1) * 16) * S_LEN + kk, S_LEN);
#pragma unroll
      for (int j = 0; j < 8; ++j) o[t][j] *= cj[j];
      o[t] = wmma_bf16(pa, bv, o[t]);
      bv = nbv;
    }
  }

  float lj[8];
#pragma unroll
  for (int j = 0; j < 8; ++j) lj[j] = 1.0f / __shfl(lrun, hb8 + j, 32);
#pragma unroll
  for (int t = 0; t < 8; ++t) {
#pragma unroll
    for (int j = 0; j < 8; ++j) {
      const int srow = q0 + hb8 + j;
      const int col = h * HD + t * 16 + r;
      O[(long)(b * S_LEN + srow) * D_DIM + col] = (bf16_t)(o[t][j] * lj[j]);
    }
  }
}

// ---------------- host-side orchestration ----------------
extern "C" void kernel_launch(void* const* d_in, const int* in_sizes, int n_in,
                              void* d_out, int out_size, void* d_ws, size_t ws_size,
                              hipStream_t stream) {
  const float* x  = (const float*)d_in[0];
  const float* Wq = (const float*)d_in[1];
  const float* Wk = (const float*)d_in[2];
  const float* Wv = (const float*)d_in[3];
  const float* Wo = (const float*)d_in[4];
  // d_in[5] = attn_mask (standard causal; applied analytically in-kernel)

  char* ws = (char*)d_ws;
  const size_t MB = (size_t)1 << 20;
  bf16_t* xb  = (bf16_t*)(ws + 0 * MB);   // 16 MB  [B,S,D]
  bf16_t* WqB = (bf16_t*)(ws + 16 * MB);  //  8 MB
  bf16_t* WkB = (bf16_t*)(ws + 24 * MB);  //  8 MB
  bf16_t* WvB = (bf16_t*)(ws + 32 * MB);  //  8 MB
  bf16_t* WoB = (bf16_t*)(ws + 40 * MB);  //  8 MB
  bf16_t* Qb  = (bf16_t*)(ws + 48 * MB);  // 16 MB  [B,H,S,hd]
  bf16_t* Kb  = (bf16_t*)(ws + 64 * MB);  // 16 MB  [B,H,S,hd]
  bf16_t* Vt  = (bf16_t*)(ws + 80 * MB);  // 16 MB  [B,H,hd,S]
  bf16_t* Ab  = (bf16_t*)(ws + 96 * MB);  // 16 MB  [B,S,D]

  const int nx = BATCH * S_LEN * D_DIM;  // 8388608
  const int nw = D_DIM * D_DIM;          // 4194304
  cvt_kernel<<<(nx + 255) / 256, 256, 0, stream>>>(x, xb, nx);
  cvt_kernel<<<(nw + 255) / 256, 256, 0, stream>>>(Wq, WqB, nw);
  cvt_kernel<<<(nw + 255) / 256, 256, 0, stream>>>(Wk, WkB, nw);
  cvt_kernel<<<(nw + 255) / 256, 256, 0, stream>>>(Wv, WvB, nw);
  cvt_kernel<<<(nw + 255) / 256, 256, 0, stream>>>(Wo, WoB, nw);

  // (M/BM)*(N/BN) = 16*16 = 256 blocks of 256 threads
  gemm_lds<<<256, 256, 0, stream>>>(xb, WqB, Qb, nullptr, 0);
  gemm_lds<<<256, 256, 0, stream>>>(xb, WkB, Kb, nullptr, 1);
  gemm_lds<<<256, 256, 0, stream>>>(xb, WvB, Vt, nullptr, 2);

  const int npairs = BATCH * NHEAD * S_LEN * (HD / 2);  // 4194304
  rope_kernel<<<(npairs + 255) / 256, 256, 0, stream>>>(Qb, npairs);
  rope_kernel<<<(npairs + 255) / 256, 256, 0, stream>>>(Kb, npairs);

  // (B*H)*(S/16) = 32*128 = 4096 waves -> 1024 blocks of 4 waves
  flash_kernel<<<1024, 128, 0, stream>>>(Qb, Kb, Vt, Ab);

  gemm_lds<<<256, 256, 0, stream>>>(Ab, WoB, nullptr, (float*)d_out, 3);
}
